// GPT_3212635537784
// MI455X (gfx1250) — compile-verified, matrix operand order checked by reading, not run on hardware
//
#include <hip/hip_runtime.h>

// ---------------------------------------------------------------------------
// GPT-2 small forward on gfx1250 (wave32, WMMA f32<-f16 16x16x32)
// - weights consumed as f32 directly (converted in-register while staging)
// - GEMM: 64x64 C tile per block (4 waves), B panel staged through
//   double-buffered LDS; f16 operands staged with global_load_async_to_lds
//   (ASYNCcnt) so the copy of k-step s+1 overlaps the WMMAs of step s.
// ---------------------------------------------------------------------------

typedef __attribute__((ext_vector_type(16))) _Float16 v16h;
typedef __attribute__((ext_vector_type(8)))  _Float16 v8h;
typedef __attribute__((ext_vector_type(8)))  float    v8f;

static constexpr int VOC = 50257;
static constexpr int TT  = 1024;
static constexpr int LL  = 12;
static constexpr int HH  = 12;
static constexpr int EE  = 768;
static constexpr int HS  = 64;
static constexpr int BB  = 2;
static constexpr int MT  = BB * TT;   // 2048 tokens
static constexpr int FF  = 4 * EE;    // 3072

static constexpr int TPB = 256;

// ---------------------------------------------------------------------------
// elementwise helpers
// ---------------------------------------------------------------------------
__global__ void k_gelu(const float* __restrict__ src,
                       _Float16* __restrict__ dst, int n) {
  int i = blockIdx.x * blockDim.x + threadIdx.x;
  if (i < n) {
    float x = src[i];
    float g = 0.5f * x * (1.f + erff(x * 0.70710678118654752f));
    dst[i] = (_Float16)g;
  }
}

__global__ void k_embed(const int* __restrict__ x,
                        const float* __restrict__ tok,
                        const float* __restrict__ pos,
                        float* __restrict__ h, _Float16* __restrict__ hf) {
  int row = blockIdx.x;            // b*T + t
  int t   = row % TT;
  int id  = x[row];
  for (int e = threadIdx.x; e < EE; e += blockDim.x) {
    float v = tok[(size_t)id * EE + e] + pos[(size_t)t * EE + e];
    h [(size_t)row * EE + e] = v;
    hf[(size_t)row * EE + e] = (_Float16)v;
  }
}

// v[b*T+t][h*64+d] (f16) -> vT[((b*H+h)*64+d)*T + t] (f16)
__global__ void k_transpose_v(const _Float16* __restrict__ v,
                              _Float16* __restrict__ vt) {
  int i = blockIdx.x * blockDim.x + threadIdx.x;
  if (i >= MT * EE) return;
  int row = i / EE, c = i % EE;
  int b = row / TT, t = row % TT;
  int h = c / HS,  d = c % HS;
  vt[(((size_t)(b * HH + h) * HS + d) * TT) + t] = v[i];
}

// ---------------------------------------------------------------------------
// residual add + LayerNorm (row = one token, 256 thr, 3 elems/thr)
// ---------------------------------------------------------------------------
__global__ void k_add_ln(const float* __restrict__ res,
                         const float* __restrict__ add,
                         const float* __restrict__ w,
                         const float* __restrict__ b,
                         float* __restrict__ hout,
                         _Float16* __restrict__ hfout) {
  __shared__ float red[TPB];
  constexpr int PER = EE / TPB;       // 3
  int row = blockIdx.x;
  float vals[PER];
  float s = 0.f;
#pragma unroll
  for (int i = 0; i < PER; i++) {
    int e = threadIdx.x + i * TPB;
    float v = res[(size_t)row * EE + e];
    if (add) v += add[(size_t)row * EE + e];
    vals[i] = v; s += v;
  }
  red[threadIdx.x] = s; __syncthreads();
  for (int off = TPB / 2; off > 0; off >>= 1) {
    if (threadIdx.x < off) red[threadIdx.x] += red[threadIdx.x + off];
    __syncthreads();
  }
  float mean = red[0] / EE; __syncthreads();
  float vs = 0.f;
#pragma unroll
  for (int i = 0; i < PER; i++) { float d = vals[i] - mean; vs += d * d; }
  red[threadIdx.x] = vs; __syncthreads();
  for (int off = TPB / 2; off > 0; off >>= 1) {
    if (threadIdx.x < off) red[threadIdx.x] += red[threadIdx.x + off];
    __syncthreads();
  }
  float rstd = rsqrtf(red[0] / EE + 1e-5f);
#pragma unroll
  for (int i = 0; i < PER; i++) {
    int e = threadIdx.x + i * TPB;
    float o = (vals[i] - mean) * rstd * w[e] + b[e];
    hout [(size_t)row * EE + e] = o;
    hfout[(size_t)row * EE + e] = (_Float16)o;
  }
}

// ---------------------------------------------------------------------------
// causal softmax over axis t: rows are (b,h,s); valid cols t <= s
// f32 scores in, f16 probs out (masked cols -> 0)
// ---------------------------------------------------------------------------
__global__ void k_softmax(const float* __restrict__ scores,
                          _Float16* __restrict__ probs) {
  __shared__ float red[TPB];
  constexpr int PER = TT / TPB;       // 4
  int row = blockIdx.x;               // (b*H + h)*T + s
  int s   = row % TT;
  const float* src = scores + (size_t)row * TT;
  _Float16*    dst = probs  + (size_t)row * TT;

  float mx = -3.4e38f;
#pragma unroll
  for (int i = 0; i < PER; i++) {
    int t = threadIdx.x + i * TPB;
    if (t <= s) mx = fmaxf(mx, src[t]);
  }
  red[threadIdx.x] = mx; __syncthreads();
  for (int off = TPB / 2; off > 0; off >>= 1) {
    if (threadIdx.x < off)
      red[threadIdx.x] = fmaxf(red[threadIdx.x], red[threadIdx.x + off]);
    __syncthreads();
  }
  mx = red[0]; __syncthreads();

  float ev[PER];
  float sum = 0.f;
#pragma unroll
  for (int i = 0; i < PER; i++) {
    int t = threadIdx.x + i * TPB;
    float e = 0.f;
    if (t <= s) e = expf(src[t] - mx);
    ev[i] = e; sum += e;
  }
  red[threadIdx.x] = sum; __syncthreads();
  for (int off = TPB / 2; off > 0; off >>= 1) {
    if (threadIdx.x < off) red[threadIdx.x] += red[threadIdx.x + off];
    __syncthreads();
  }
  float inv = 1.f / red[0];
#pragma unroll
  for (int i = 0; i < PER; i++) {
    int t = threadIdx.x + i * TPB;
    dst[t] = (_Float16)(ev[i] * inv);
  }
}

// ---------------------------------------------------------------------------
// fragment loader (CDNA5 ISA 7.12.2 layout for 16-bit A 16x32 / B 32x16):
//   lanes 0-15  : row = lane,    K offsets {0..7, 16..23}
//   lanes 16-31 : row = lane-16, K offsets {8..15, 24..31}
// p points at (row, koff) of the operand; chunks at +0 and +16 elements.
// ---------------------------------------------------------------------------
__device__ __forceinline__ v16h load_frag(const _Float16* p) {
  v8h a0 = *(const v8h*)p;
  v8h a1 = *(const v8h*)(p + 16);
  return __builtin_shufflevector(a0, a1, 0,1,2,3,4,5,6,7,8,9,10,11,12,13,14,15);
}

// ---------------------------------------------------------------------------
// B-panel staging into LDS: one thread moves 16 consecutive k-elements of one
// B row (32 bytes of f16). f32 source: load + v_cvt_pk + ds_store.
// f16 source: global_load_async_to_lds (2 x b128; INST_OFFSET applies to both
// the global and LDS addresses, per ISA 15.18.3), tracked by ASYNCcnt.
// ---------------------------------------------------------------------------
__device__ __forceinline__ void stage_b(const float* g, _Float16* s) {
  v8f f0 = *(const v8f*)g;
  v8f f1 = *(const v8f*)(g + 8);
  *(v8h*)s       = __builtin_convertvector(f0, v8h);
  *(v8h*)(s + 8) = __builtin_convertvector(f1, v8h);
}
__device__ __forceinline__ void stage_b(const _Float16* g, _Float16* s) {
  unsigned int lds = (unsigned int)(unsigned long long)s;  // addr[31:0] = LDS offset
  asm volatile("global_load_async_to_lds_b128 %0, %1, off\n\t"
               "global_load_async_to_lds_b128 %0, %1, off offset:16"
               :: "v"(lds), "v"(g) : "memory");
}
__device__ __forceinline__ void async_fence() {
  asm volatile("s_wait_asynccnt 0x0" ::: "memory");
}

// ---------------------------------------------------------------------------
// WMMA GEMM: C[M,N] = alpha * A[M,K] * W[N,K]^T + bias[N]
// A f16 row-major; W f16 or f32 row-major. Block = 4 waves = 64x64 C tile
// (wave w owns rows mBase..mBase+15, all waves share the 64-col B panel).
// B panel staged through double-buffered LDS, one 32-deep k-step ahead.
// Requires M % 64 == 0, K % 32 == 0; N arbitrary (clamped loads, guarded
// stores; WMMA always executes with full EXEC).
// ---------------------------------------------------------------------------
template <typename TB>
__global__ void k_wmma_gemm(const _Float16* __restrict__ A, long long strA, int lda,
                            const TB* __restrict__ Bw, long long strB, int ldb,
                            float* __restrict__ C, _Float16* __restrict__ Ch,
                            long long strC, int ldc,
                            const float* __restrict__ bias, float alpha,
                            int M, int N, int K) {
  constexpr int LSTR = 40;                       // halfs per LDS row (32 + 8 pad)
  __shared__ _Float16 sPan[2][64 * LSTR];

  const int lane = threadIdx.x;                  // 0..31
  const int wave = threadIdx.y;                  // 0..3 -> M subtile
  const int tid  = wave * 32 + lane;
  const long long nBase = (long long)blockIdx.x * 64;
  const int mBase = blockIdx.y * 64 + wave * 16;

  const int z = blockIdx.z;
  A  += (size_t)z * strA;
  Bw += (size_t)z * strB;
  const size_t cOff = (size_t)z * strC;

  const int hi = lane >> 4, l15 = lane & 15, koff = hi * 8;
  const _Float16* aptr = A + (size_t)(mBase + l15) * lda + koff;

  // staging assignment: thread -> (panel row, which 16-elem half of k-chunk)
  const int srow  = tid >> 1;                    // 0..63
  const int shalf = (tid & 1) * 16;              // 0 | 16
  long long gRow = nBase + srow; if (gRow > N - 1) gRow = N - 1;
  const TB* bsrc = Bw + (size_t)gRow * ldb + shalf;
  const int sOff = srow * LSTR + shalf;

  const int S = K / 32;
  stage_b(bsrc, &sPan[0][sOff]);

  v8f acc[4] = {};
  for (int s = 0; s < S; s++) {
    async_fence();                               // this wave's async copies done
    __syncthreads();                             // panel s visible to all waves
    if (s + 1 < S)
      stage_b(bsrc + (size_t)(s + 1) * 32, &sPan[(s + 1) & 1][sOff]);
    const _Float16* pan = &sPan[s & 1][0];
    v16h av = load_frag(aptr + (size_t)s * 32);
#pragma unroll
    for (int j = 0; j < 4; j++) {
      v16h bv = load_frag(pan + (j * 16 + l15) * LSTR + koff);
      acc[j] = __builtin_amdgcn_wmma_f32_16x16x32_f16(false, av, false, bv,
                                                      (short)0, acc[j],
                                                      false, false);
    }
    __syncthreads();                             // panel s free for re-staging
  }

#pragma unroll
  for (int j = 0; j < 4; j++) {
    long long col = nBase + j * 16 + l15;
    if (col < N) {
      float bv = bias ? bias[col] : 0.f;
#pragma unroll
      for (int i = 0; i < 8; i++) {
        int r = mBase + i + hi * 8;
        float v = acc[j][i] * alpha + bv;
        size_t idx = cOff + (size_t)r * ldc + (size_t)col;
        if (C)  C[idx]  = v;
        if (Ch) Ch[idx] = (_Float16)v;
      }
    }
  }
}

// ---------------------------------------------------------------------------
// host orchestration
// ---------------------------------------------------------------------------
static inline size_t alignUp(size_t v) { return (v + 255) & ~(size_t)255; }

template <typename TB>
static void launch_gemm(hipStream_t stream,
                        const _Float16* A, long long sA, int lda,
                        const TB* Bw, long long sB, int ldb,
                        float* C, _Float16* Ch, long long sC, int ldc,
                        const float* bias, float alpha,
                        int M, int N, int K, int batch) {
  dim3 blk(32, 4, 1);
  dim3 grd((N + 63) / 64, M / 64, batch);
  k_wmma_gemm<TB><<<grd, blk, 0, stream>>>(A, sA, lda, Bw, sB, ldb,
                                           C, Ch, sC, ldc, bias, alpha, M, N, K);
}

extern "C" void kernel_launch(void* const* d_in, const int* in_sizes, int n_in,
                              void* d_out, int out_size, void* d_ws, size_t ws_size,
                              hipStream_t stream) {
  const int*   x       = (const int*)  d_in[0];
  const float* tok_emb = (const float*)d_in[1];
  const float* pos_emb = (const float*)d_in[2];
  const float* qW      = (const float*)d_in[3];
  const float* kW      = (const float*)d_in[4];
  const float* vW      = (const float*)d_in[5];
  const float* ln1_w   = (const float*)d_in[6];
  const float* ln1_b   = (const float*)d_in[7];
  const float* fc1_w   = (const float*)d_in[8];
  const float* fc1_b   = (const float*)d_in[9];
  const float* fc2_w   = (const float*)d_in[10];
  const float* fc2_b   = (const float*)d_in[11];
  const float* ln2_w   = (const float*)d_in[12];
  const float* ln2_b   = (const float*)d_in[13];
  const float* lnf_w   = (const float*)d_in[14];
  const float* lnf_b   = (const float*)d_in[15];
  const float* lm_w    = (const float*)d_in[16];
  float* out = (float*)d_out;

  char* ws = (char*)d_ws;
  size_t off = 0;
  auto carve = [&](size_t bytes) { size_t o = off; off = alignUp(off + bytes); return o; };

  float*    h      = (float*)   (ws + carve((size_t)MT * EE * 4));
  _Float16* hf     = (_Float16*)(ws + carve((size_t)MT * EE * 2));
  _Float16* qh     = (_Float16*)(ws + carve((size_t)MT * EE * 2));
  _Float16* kh     = (_Float16*)(ws + carve((size_t)MT * EE * 2));
  _Float16* vh     = (_Float16*)(ws + carve((size_t)MT * EE * 2));
  _Float16* vth    = (_Float16*)(ws + carve((size_t)MT * EE * 2));
  float*    scores = (float*)   (ws + carve((size_t)BB * HH * TT * TT * 4));
  _Float16* probs  = (_Float16*)(ws + carve((size_t)BB * HH * TT * TT * 2));
  float*    aout   = (float*)   (ws + carve((size_t)MT * EE * 4));
  float*    mlp1   = (float*)   (ws + carve((size_t)MT * FF * 4));
  _Float16* mlp1h  = (_Float16*)(ws + carve((size_t)MT * FF * 2));
  float*    mlpout = (float*)   (ws + carve((size_t)MT * EE * 4));
  (void)ws_size; (void)in_sizes; (void)n_in; (void)out_size;

  const float attScale = 0.03608439182435161f;   // 1/sqrt(768)

  // embeddings (f32 + f16 copies)
  k_embed<<<MT, TPB, 0, stream>>>(x, tok_emb, pos_emb, h, hf);

  for (int l = 0; l < LL; l++) {
    const float* qWl  = qW    + (size_t)l * EE * EE;
    const float* kWl  = kW    + (size_t)l * EE * EE;
    const float* vWl  = vW    + (size_t)l * EE * EE;
    const float* f1Wl = fc1_w + (size_t)l * FF * EE;
    const float* f2Wl = fc2_w + (size_t)l * EE * FF;

    // QKV projections (f32 weights consumed directly, f16 outputs)
    launch_gemm<float>(stream, hf, 0, EE, qWl, 0, EE, nullptr, qh, 0, EE,
                       nullptr, 1.f, MT, EE, EE, 1);
    launch_gemm<float>(stream, hf, 0, EE, kWl, 0, EE, nullptr, kh, 0, EE,
                       nullptr, 1.f, MT, EE, EE, 1);
    launch_gemm<float>(stream, hf, 0, EE, vWl, 0, EE, nullptr, vh, 0, EE,
                       nullptr, 1.f, MT, EE, EE, 1);
    k_transpose_v<<<(MT * EE + 255) / 256, 256, 0, stream>>>(vh, vth);

    // scores[b,h,s,t] = scale * k[b,s,h,:] . q[b,t,h,:]  (batched over h)
    for (int b = 0; b < BB; b++) {
      launch_gemm<_Float16>(stream,
           kh + (size_t)b * TT * EE, HS, EE,
           qh + (size_t)b * TT * EE, HS, EE,
           scores + (size_t)b * HH * TT * TT, nullptr, (long long)TT * TT, TT,
           nullptr, attScale, TT, TT, HS, HH);
    }
    k_softmax<<<BB * HH * TT, TPB, 0, stream>>>(scores, probs);

    // a[b,s,h,d] = sum_t probs[b,h,s,t] * v[b,t,h,d]
    for (int b = 0; b < BB; b++) {
      launch_gemm<_Float16>(stream,
           probs + (size_t)b * HH * TT * TT, (long long)TT * TT, TT,
           vth   + (size_t)b * HH * HS * TT, (long long)HS * TT, TT,
           aout  + (size_t)b * TT * EE, nullptr, HS, EE,
           nullptr, 1.f, TT, HS, TT, HH);
    }

    // h = LN1(h + a)
    k_add_ln<<<MT, TPB, 0, stream>>>(h, aout, ln1_w + (size_t)l * EE,
                                     ln1_b + (size_t)l * EE, h, hf);

    // MLP
    launch_gemm<float>(stream, hf, 0, EE, f1Wl, 0, EE, mlp1, nullptr, 0, FF,
                       fc1_b + (size_t)l * FF, 1.f, MT, FF, EE, 1);
    k_gelu<<<(MT * FF + 255) / 256, 256, 0, stream>>>(mlp1, mlp1h, MT * FF);
    launch_gemm<float>(stream, mlp1h, 0, FF, f2Wl, 0, FF, mlpout, nullptr, 0, EE,
                       fc2_b + (size_t)l * EE, 1.f, MT, EE, FF, 1);

    // h = LN2(h + m)
    k_add_ln<<<MT, TPB, 0, stream>>>(h, mlpout, ln2_w + (size_t)l * EE,
                                     ln2_b + (size_t)l * EE, h, hf);
  }

  // final LN + logits (lm_w f32 consumed directly)
  k_add_ln<<<MT, TPB, 0, stream>>>(h, nullptr, lnf_w, lnf_b, h, hf);
  launch_gemm<float>(stream, hf, 0, EE, lm_w, 0, EE, out, nullptr, 0, VOC,
                     nullptr, 1.f, MT, VOC, EE, 1);
}
